// WindowAttention_83047487635843
// MI455X (gfx1250) — matrix operand order repaired and checked by reading, not compiled, not run on hardware
//
#include <hip/hip_runtime.h>
#include <hip/hip_bf16.h>

typedef __attribute__((ext_vector_type(16))) __bf16 v16bf;
typedef __attribute__((ext_vector_type(8)))  float  v8f;

#define DIMC   256
#define NTOK   64
#define NHEAD  8
#define HDIM   32
#define QSCALE 0.17677669529663687f   /* 1/sqrt(32) */

/* ---- LDS layout (units: __bf16 elements). Row strides are multiples of 8
   (16B) for aligned b128 access, padded for conflict-free bank rotation.
   The softmax-prob strip reuses the (dead-by-then) per-head K region.     ---- */
#define XS_OFF   0            /* x as bf16: 64 x 264 ; reused as per-head out */
#define XS_LD    264
#define QS_OFF   16896        /* Q:  8 heads x 64 tok x 40 (dim 32 + pad 8)   */
#define QK_LD    40
#define KS_OFF   37376        /* K:  same shape as Q; per-head region reused
                                 as the 16x72 softmax-prob strip in stage C  */
#define VT_OFF   57856        /* V^T: 8 heads x 32 dim x 72 (tok 64 + pad 8)  */
#define VT_LD    72
#define PS_LD    72
#define LDS_ELEMS 76288       /* *2 bytes = 152576 B -> 2 workgroups per WGP  */

/* workspace layout (bytes) */
#define WS_WQKV   0                       /* 768*256 bf16 = 393216 B */
#define WS_WPROJ  393216                  /* 256*256 bf16 = 131072 B */
#define WS_BIAS   (393216 + 131072)       /* 8*4*4*32*8 f32 = 131072 B */

struct U128x2 { uint4 lo, hi; };
struct F32x8  { float4 lo, hi; };
struct BF8    { __bf16 v[8]; };           /* 16 B */

static __device__ __forceinline__ v16bf mkv16(uint4 lo, uint4 hi) {
  U128x2 t{lo, hi};
  return __builtin_bit_cast(v16bf, t);
}
static __device__ __forceinline__ v8f mkv8f(float4 lo, float4 hi) {
  F32x8 t{lo, hi};
  return __builtin_bit_cast(v8f, t);
}

/* A-matrix 16x32 bf16 layout: lanes 0-15 row m hold K {0..7,16..23};
   lanes 16-31 hold K {8..15,24..31}. Caller passes e0 = row*ld + k0 + hi*8. */
static __device__ __forceinline__ v16bf load_a(const __bf16* s, int e0) {
  uint4 c0 = *(const uint4*)(s + e0);
  uint4 c1 = *(const uint4*)(s + e0 + 16);
  return mkv16(c0, c1);
}
/* B-matrix 32x16 bf16 layout: lane n (col) holds K {0..7,16..23} (lanes 0-15)
   or K {8..15,24..31} (lanes 16-31). Caller passes e0 = col_base + hi*8. */
static __device__ __forceinline__ v16bf load_b_lds(const __bf16* s, int e0) {
  uint4 c0 = *(const uint4*)(s + e0);
  uint4 c1 = *(const uint4*)(s + e0 + 16);
  return mkv16(c0, c1);
}
static __device__ __forceinline__ v16bf load_b_glb(const __bf16* g, int e0) {
  uint4 c0 = *(const uint4*)(g + e0);
  uint4 c1 = *(const uint4*)(g + e0 + 16);
  return mkv16(c0, c1);
}

static __device__ __forceinline__ v8f wmma_bf16(v16bf a, v16bf b, v8f c) {
  return __builtin_amdgcn_wmma_f32_16x16x32_bf16(
      /*neg_a=*/false, a, /*neg_b=*/false, b,
      /*c_mod=*/(short)0, c, /*reuse_a=*/false, /*reuse_b=*/false);
}

/* ------------------------------------------------------------------------- */
/* Pre-pass: bf16 weight conversion + bias rearranged into WMMA C-tile order */
/* biasC[(((h*4+i)*4+j)*32+lane)*8 + r] = rpb[rpi[m][n]][h],
   m = i*16 + r + (lane>=16)*8 , n = j*16 + lane%16                          */
/* ------------------------------------------------------------------------- */
__global__ void prep_kernel(const float* __restrict__ wqkv,
                            const float* __restrict__ wproj,
                            const float* __restrict__ rpb,
                            const int*   __restrict__ rpi,
                            __bf16* __restrict__ wqkv_bf,
                            __bf16* __restrict__ wproj_bf,
                            float*  __restrict__ biasC) {
  const int NQ = 768 * 256, NP = 256 * 256, NB = 8 * 4 * 4 * 32 * 8;
  int idx = blockIdx.x * blockDim.x + threadIdx.x;
  for (int e = idx; e < NQ + NP + NB; e += gridDim.x * blockDim.x) {
    if (e < NQ) {
      wqkv_bf[e] = (__bf16)wqkv[e];
    } else if (e < NQ + NP) {
      wproj_bf[e - NQ] = (__bf16)wproj[e - NQ];
    } else {
      int t = e - NQ - NP;
      int r = t & 7, l = (t >> 3) & 31, j = (t >> 8) & 3;
      int i = (t >> 10) & 3, h = t >> 12;
      int m = i * 16 + r + (l >> 4) * 8;
      int n = j * 16 + (l & 15);
      biasC[t] = rpb[rpi[m * 64 + n] * 8 + h];
    }
  }
}

/* ------------------------------------------------------------------------- */
/* Fused window attention: one workgroup (8 waves) per window                */
/* ------------------------------------------------------------------------- */
__global__ void __launch_bounds__(256)
win_attn_kernel(const float* __restrict__ x,
                const float* __restrict__ b_qkv,
                const float* __restrict__ b_proj,
                const __bf16* __restrict__ wqkv_bf,
                const __bf16* __restrict__ wproj_bf,
                const float* __restrict__ biasC,
                float* __restrict__ out) {
  extern __shared__ __align__(16) char smem_raw[];
  __bf16* sm = (__bf16*)smem_raw;

  const int  tid  = threadIdx.x;
  /* wave id as a proven-scalar: keeps all per-wave routing in SALU code */
  const int  w    = __builtin_amdgcn_readfirstlane(tid >> 5);
  const int  lane = tid & 31;
  const int  ln16 = lane & 15;
  const int  hi   = lane >> 4;      /* half-wave: 0 or 1 */
  const long b    = blockIdx.x;

  /* ---- Stage A: x[b] (f32) -> xs (bf16): 8 elems/iter, one b128 store ---- */
  {
    const float* xb = x + b * (NTOK * DIMC);
    for (int e = tid * 8; e < NTOK * DIMC; e += 256 * 8) {
      float4 v0 = *(const float4*)(xb + e);
      float4 v1 = *(const float4*)(xb + e + 4);
      BF8 t;
      t.v[0] = (__bf16)v0.x; t.v[1] = (__bf16)v0.y;
      t.v[2] = (__bf16)v0.z; t.v[3] = (__bf16)v0.w;
      t.v[4] = (__bf16)v1.x; t.v[5] = (__bf16)v1.y;
      t.v[6] = (__bf16)v1.z; t.v[7] = (__bf16)v1.w;
      *(uint4*)(sm + XS_OFF + (e >> 8) * XS_LD + (e & 255)) =
          __builtin_bit_cast(uint4, t);
    }
  }
  __syncthreads();

  /* ---- Stage B: qkv = xs @ wqkv^T + b_qkv -> Q(scaled)/K/V^T in LDS ---- */
  for (int jj = 0; jj < 6; ++jj) {
    const int j    = w * 6 + jj;        /* N-tile 0..47 over 768 cols (SGPR) */
    const int col0 = j * 16;
    const int mat  = col0 >> 8;         /* 0=q 1=k 2=v  (scalar) */
    const int head = (col0 & 255) >> 5; /* scalar */
    const int dim0 = col0 & 31;         /* 0 or 16 (scalar) */
    const float bias = b_qkv[col0 + ln16];
    const int brow = (col0 + ln16) * DIMC;
    for (int i = 0; i < 4; ++i) {
      v8f acc = {};
      for (int k = 0; k < 8; ++k) {
        v16bf a  = load_a(sm + XS_OFF, (i * 16 + ln16) * XS_LD + k * 32 + hi * 8);
        v16bf bm = load_b_glb(wqkv_bf, brow + k * 32 + hi * 8);
        acc = wmma_bf16(a, bm, acc);
      }
      const int tb = i * 16 + hi * 8;   /* first token of this lane's rows */
      const int d  = dim0 + ln16;       /* dim within head (per lane) */
      if (mat == 2) {
        /* V^T: the 8 C elements are consecutive tokens -> one b128 store */
        BF8 t;
        for (int r = 0; r < 8; ++r) t.v[r] = (__bf16)(acc[r] + bias);
        *(uint4*)(sm + VT_OFF + (head * 32 + d) * VT_LD + tb) =
            __builtin_bit_cast(uint4, t);
      } else if (mat == 0) {
        for (int r = 0; r < 8; ++r)
          sm[QS_OFF + (head * 64 + tb + r) * QK_LD + d] =
              (__bf16)((acc[r] + bias) * QSCALE);
      } else {
        for (int r = 0; r < 8; ++r)
          sm[KS_OFF + (head * 64 + tb + r) * QK_LD + d] = (__bf16)(acc[r] + bias);
      }
    }
  }
  __syncthreads();

  /* ---- Stage C: per-head attention (wave w == head w) ---- */
  {
    const int h = w;
    const __bf16* qh  = sm + QS_OFF + h * 64 * QK_LD;
    const __bf16* kh  = sm + KS_OFF + h * 64 * QK_LD;
    const __bf16* vh  = sm + VT_OFF + h * 32 * VT_LD;
    /* K region for this head is dead once kb[] is in registers: reuse it
       as the wave-private prob strip (16 x 72 = 1152 elems < 64*40 = 2560) */
    __bf16* psw = sm + KS_OFF + h * 64 * QK_LD;

    v16bf kb[4], vb[2][2];
    for (int jt = 0; jt < 4; ++jt)
      kb[jt] = load_b_lds(kh, (jt * 16 + ln16) * QK_LD + hi * 8);
    for (int kt = 0; kt < 2; ++kt)
      for (int nt = 0; nt < 2; ++nt)
        vb[kt][nt] = load_b_lds(vh, (nt * 16 + ln16) * VT_LD + kt * 32 + hi * 8);

    for (int i = 0; i < 4; ++i) {
      v16bf qa = load_a(qh, (i * 16 + ln16) * QK_LD + hi * 8);
      v8f S[4];
      for (int jt = 0; jt < 4; ++jt) {
        v8f z = {};
        v8f c = wmma_bf16(qa, kb[jt], z);   /* K=32 = one WMMA per tile */
        const float* bp = biasC + (((h * 4 + i) * 4 + jt) * 32 + lane) * 8;
        float4 b0 = *(const float4*)bp;
        float4 b1 = *(const float4*)(bp + 4);
        S[jt] = c + mkv8f(b0, b1);
      }
      /* softmax over the 64 columns of each row; rows live per-(reg, half) */
      float mx[8], inv[8];
      for (int r = 0; r < 8; ++r) {
        float m = fmaxf(fmaxf(S[0][r], S[1][r]), fmaxf(S[2][r], S[3][r]));
        for (int d = 1; d < 16; d <<= 1) m = fmaxf(m, __shfl_xor(m, d, 16));
        mx[r] = m;
      }
      for (int jt = 0; jt < 4; ++jt)
        for (int r = 0; r < 8; ++r) S[jt][r] = __expf(S[jt][r] - mx[r]);
      for (int r = 0; r < 8; ++r) {
        float s = S[0][r] + S[1][r] + S[2][r] + S[3][r];
        for (int d = 1; d < 16; d <<= 1) s += __shfl_xor(s, d, 16);
        inv[r] = 1.0f / s;
      }
      /* re-stripe probs into A-layout via wave-private LDS strip */
      for (int jt = 0; jt < 4; ++jt)
        for (int r = 0; r < 8; ++r)
          psw[(r + hi * 8) * PS_LD + jt * 16 + ln16] = (__bf16)(S[jt][r] * inv[r]);

      v16bf pa0 = load_a(psw, ln16 * PS_LD + hi * 8);
      v16bf pa1 = load_a(psw, ln16 * PS_LD + 32 + hi * 8);
      for (int nt = 0; nt < 2; ++nt) {
        v8f z = {};
        v8f o = wmma_bf16(pa0, vb[0][nt], z);
        o = wmma_bf16(pa1, vb[1][nt], o);
        for (int r = 0; r < 8; ++r)   /* per-head output -> reuse xs buffer */
          sm[XS_OFF + (i * 16 + hi * 8 + r) * XS_LD + h * HDIM + nt * 16 + ln16] =
              (__bf16)o[r];
      }
    }
  }
  __syncthreads();

  /* ---- Stage E: out = os @ wproj^T + b_proj (straight to global f32) ---- */
  {
    float* outb = out + b * (NTOK * DIMC);
    for (int jj = 0; jj < 2; ++jj) {
      const int jn   = w * 2 + jj;     /* N-tile 0..15 over 256 cols (SGPR) */
      const int col0 = jn * 16;
      const float bias = b_proj[col0 + ln16];
      const int brow = (col0 + ln16) * DIMC;
      for (int i = 0; i < 4; ++i) {
        v8f acc = {};
        for (int k = 0; k < 8; ++k) {
          v16bf a  = load_a(sm + XS_OFF, (i * 16 + ln16) * XS_LD + k * 32 + hi * 8);
          v16bf bm = load_b_glb(wproj_bf, brow + k * 32 + hi * 8);
          acc = wmma_bf16(a, bm, acc);
        }
        for (int r = 0; r < 8; ++r)
          outb[(i * 16 + hi * 8 + r) * DIMC + col0 + ln16] = acc[r] + bias;
      }
    }
  }
}

/* ------------------------------------------------------------------------- */
extern "C" void kernel_launch(void* const* d_in, const int* in_sizes, int n_in,
                              void* d_out, int out_size, void* d_ws, size_t ws_size,
                              hipStream_t stream) {
  const float* x     = (const float*)d_in[0];
  const float* wqkv  = (const float*)d_in[1];
  const float* bqkv  = (const float*)d_in[2];
  const float* wproj = (const float*)d_in[3];
  const float* bproj = (const float*)d_in[4];
  const float* rpb   = (const float*)d_in[5];
  const int*   rpi   = (const int*)d_in[6];
  float*       outp  = (float*)d_out;

  char* ws = (char*)d_ws;
  __bf16* wqkv_bf  = (__bf16*)(ws + WS_WQKV);
  __bf16* wproj_bf = (__bf16*)(ws + WS_WPROJ);
  float*  biasC    = (float*)(ws + WS_BIAS);

  const int n_windows = in_sizes[0] / (NTOK * DIMC);

  prep_kernel<<<288, 256, 0, stream>>>(wqkv, wproj, rpb, rpi,
                                       wqkv_bf, wproj_bf, biasC);

  const size_t lds_bytes = (size_t)LDS_ELEMS * sizeof(__bf16);
  win_attn_kernel<<<n_windows, 256, lds_bytes, stream>>>(
      x, bqkv, bproj, wqkv_bf, wproj_bf, biasC, outp);
}
